// VisibilityGNN_39702677684436
// MI455X (gfx1250) — matrix operand order repaired and checked by reading, not compiled, not run on hardware
//
#include <hip/hip_runtime.h>
#include <hip/hip_bf16.h>
#include <math.h>

// ---------------------------------------------------------------------------
// VisibilityGNN for MI455X (gfx1250): wave32 + WMMA bf16 (f32 accum).
// Dead code eliminated: hf, hi(enc), f1, c1_rb, c1_af never affect the output.
// Round 2: branch-free vectorized LDS staging (float4 loads, packed bf16
// b128/b64 LDS stores) to remove the serial per-element staging loops.
// ---------------------------------------------------------------------------

#define ND 8192
#define NI 1024
#define NF 512
#define DD 64
#define H  256
#define E_HI 262144
#define E_RB 65536
#define E_AF 131072

typedef __bf16 bf16;
typedef bf16 v16bf __attribute__((ext_vector_type(16)));
typedef bf16 v8bf  __attribute__((ext_vector_type(8)));
typedef bf16 v4bf  __attribute__((ext_vector_type(4)));
typedef float v8f  __attribute__((ext_vector_type(8)));

#define TM 128
#define TN 128
#define TK 32

__device__ __forceinline__ v8bf pack8(float4 a, float4 b) {
  return (v8bf){(bf16)a.x, (bf16)a.y, (bf16)a.z, (bf16)a.w,
                (bf16)b.x, (bf16)b.y, (bf16)b.z, (bf16)b.w};
}
__device__ __forceinline__ v4bf pack4(float a, float b, float c, float d) {
  return (v4bf){(bf16)a, (bf16)b, (bf16)c, (bf16)d};
}

// Stage a 128-col x 32-k fp32 tile TRANSPOSED into LDS as bf16 Wt[col][k].
// 256 threads: each owns a 4-col x 4-k micro-tile (exact fit, branch-free).
__device__ __forceinline__ void stage_w_tile(bf16* __restrict__ Wt,
                                             const float* __restrict__ W,
                                             int ldW, int k0, int col0, int tid)
{
  const int c4 = (tid & 31) << 2;   // 0..124
  const int kk = (tid >> 5) << 2;   // 0..28
  const float* src = &W[(size_t)(k0 + kk) * ldW + col0 + c4];
  float4 w0 = *(const float4*)(src);
  float4 w1 = *(const float4*)(src + ldW);
  float4 w2 = *(const float4*)(src + 2 * ldW);
  float4 w3 = *(const float4*)(src + 3 * ldW);
  *(v4bf*)&Wt[(c4 + 0) * TK + kk] = pack4(w0.x, w1.x, w2.x, w3.x);
  *(v4bf*)&Wt[(c4 + 1) * TK + kk] = pack4(w0.y, w1.y, w2.y, w3.y);
  *(v4bf*)&Wt[(c4 + 2) * TK + kk] = pack4(w0.z, w1.z, w2.z, w3.z);
  *(v4bf*)&Wt[(c4 + 3) * TK + kk] = pack4(w0.w, w1.w, w2.w, w3.w);
}

// ---------------------------------------------------------------------------
// Generic C = act(A[MxK] @ W[KxN] + bias) with v_wmma_f32_16x16x32_bf16.
// Block = 256 threads (8 waves). Each wave computes a 16x128 slice.
// ---------------------------------------------------------------------------
__global__ __launch_bounds__(256) void gemm_bias_act_kernel(
    const float* __restrict__ A, const float* __restrict__ W,
    const float* __restrict__ bias, float* __restrict__ C,
    int M, int N, int K, int act)
{
  __shared__ __align__(16) bf16 As[TM * TK];
  __shared__ __align__(16) bf16 Wt[TN * TK];

  const int tid  = threadIdx.x;
  const int wave = tid >> 5;
  const int lane = tid & 31;
  const int l16  = lane & 15;
  const int hi16 = lane >> 4;
  const int row0 = blockIdx.x * TM;
  const int col0 = blockIdx.y * TN;

  // A staging map: thread owns (row = tid>>1, 16-k half = (tid&1)*16)
  const int ar   = tid >> 1;
  const int ah   = (tid & 1) << 4;

  v8f acc[8] = {};

  for (int k0 = 0; k0 < K; k0 += TK) {
    { // stage A tile (fp32 -> bf16): 4x b128 loads, 2x b128 LDS stores
      const float* src = &A[(size_t)(row0 + ar) * K + k0 + ah];
      float4 a0 = *(const float4*)(src);
      float4 a1 = *(const float4*)(src + 4);
      float4 a2 = *(const float4*)(src + 8);
      float4 a3 = *(const float4*)(src + 12);
      *(v8bf*)&As[ar * TK + ah]     = pack8(a0, a1);
      *(v8bf*)&As[ar * TK + ah + 8] = pack8(a2, a3);
    }
    stage_w_tile(Wt, W, N, k0, col0, tid);
    if (k0 + TK < K) {             // global_prefetch_b8 of the next K tile
      __builtin_prefetch(&A[(size_t)(row0 + ar) * K + k0 + TK + ah], 0, 1);
      __builtin_prefetch(&W[(size_t)(k0 + TK + (tid >> 5) * 4) * N + col0 + (tid & 31) * 4], 0, 1);
    }
    __syncthreads();

    // A fragment: lanes 0-15 -> row=l16, K {0..7,16..23}; lanes 16-31 -> K {8..15,24..31}
    const bf16* arow = &As[(wave * 16 + l16) * TK + (hi16 ? 8 : 0)];
    v8bf alo = *(const v8bf*)(arow);
    v8bf ahi = *(const v8bf*)(arow + 16);
    v16bf afrag = __builtin_shufflevector(alo, ahi,
        0,1,2,3,4,5,6,7,8,9,10,11,12,13,14,15);

#pragma unroll
    for (int j = 0; j < 8; ++j) {
      // B fragment: lanes 0-15 -> col, K {0..7,16..23}; lanes 16-31 -> col, K {8..15,24..31}
      const bf16* bcol = &Wt[(j * 16 + l16) * TK + (hi16 ? 8 : 0)];
      v8bf blo = *(const v8bf*)(bcol);
      v8bf bhi = *(const v8bf*)(bcol + 16);
      v16bf bfrag = __builtin_shufflevector(blo, bhi,
          0,1,2,3,4,5,6,7,8,9,10,11,12,13,14,15);
      acc[j] = __builtin_amdgcn_wmma_f32_16x16x32_bf16(
          false, afrag, false, bfrag, (short)0, acc[j], false, false);
    }
    __syncthreads();
  }

  // epilogue: C/D layout -> VGPR r holds M=r (lanes 0-15) / M=r+8 (lanes 16-31)
#pragma unroll
  for (int j = 0; j < 8; ++j) {
    int col = col0 + j * 16 + l16;
    float bv = bias ? bias[col] : 0.0f;
#pragma unroll
    for (int r = 0; r < 8; ++r) {
      int row = row0 + wave * 16 + r + hi16 * 8;
      float v = acc[j][r] + bv;
      if (act) v = fmaxf(v, 0.0f);
      C[(size_t)row * N + col] = v;
    }
  }
}

// ---------------------------------------------------------------------------
// GCN support kernels
// ---------------------------------------------------------------------------
__global__ void deg_kernel(const int* __restrict__ ei, int E,
                           float* __restrict__ degS, float* __restrict__ degD)
{
  int e = blockIdx.x * blockDim.x + threadIdx.x;
  if (e < E) {
    atomicAdd(&degS[ei[e]], 1.0f);
    atomicAdd(&degD[ei[E + e]], 1.0f);
  }
}

__global__ void invsqrt_kernel(float* __restrict__ d, int n)
{
  int i = blockIdx.x * blockDim.x + threadIdx.x;
  if (i < n) {
    float x = d[i];
    d[i] = (x > 0.0f) ? rsqrtf(x) : 0.0f;
  }
}

// one thread per (edge, 4-feature chunk); float4 gather + 4x atomic f32 add
__global__ void scatter_kernel(const float* __restrict__ feat,
                               const int* __restrict__ ei,
                               const float* __restrict__ invS,
                               const float* __restrict__ invD,
                               float* __restrict__ out, int E)
{
  int gid = blockIdx.x * blockDim.x + threadIdx.x;
  if (gid >= E * (H / 4)) return;
  int e = gid >> 6;                 // H/4 == 64
  int c = (gid & 63) << 2;
  int s = ei[e];
  int d = ei[E + e];
  float nrm = invS[s] * invD[d];
  float4 v = *(const float4*)&feat[(size_t)s * H + c];
  float* o = &out[(size_t)d * H + c];
  atomicAdd(o + 0, v.x * nrm);
  atomicAdd(o + 1, v.y * nrm);
  atomicAdd(o + 2, v.z * nrm);
  atomicAdd(o + 3, v.w * nrm);
}

__global__ void finalize_kernel(float* __restrict__ out, const float* __restrict__ acc,
                                const float* __restrict__ b1, const float* __restrict__ b2,
                                int n, int relu_)
{
  int i = blockIdx.x * blockDim.x + threadIdx.x;
  if (i < n) {
    int c = i & (H - 1);
    float v = acc[i] + b1[c] + (b2 ? b2[c] : 0.0f);
    out[i] = relu_ ? fmaxf(v, 0.0f) : v;
  }
}

// ---------------------------------------------------------------------------
// Fused cross-product MLP:
//   out[i,f] = sigmoid( relu( relu(pi[i]+pf[f]+b1) @ w2 + b2 ) . w3 + b3 )
// Block = 8 intents x 16 fixes; wave w = intent i0+w, rows M=0..15 = fixes.
// A fragments are synthesized in registers from LDS pi/pf tiles; h1/h2 never
// touch HBM (saves ~1 GB of traffic vs materialization).
// ---------------------------------------------------------------------------
__global__ __launch_bounds__(256) void pair_mlp_kernel(
    const float* __restrict__ pi, const float* __restrict__ pf,
    const float* __restrict__ b1, const float* __restrict__ w2,
    const float* __restrict__ b2, const float* __restrict__ w3,
    const float* __restrict__ b3, float* __restrict__ out)
{
  __shared__ __align__(16) bf16 Wt2[128 * TK];   // w2 K-tile, col-major
  __shared__ __align__(16) float piS[8 * H];
  __shared__ __align__(16) float pfS[16 * H];
  __shared__ float b1S[H];
  __shared__ float b2S[128];
  __shared__ float w3S[128];

  const int tid  = threadIdx.x;
  const int wave = tid >> 5;
  const int lane = tid & 31;
  const int l16  = lane & 15;
  const int hi16 = lane >> 4;
  const int i0 = blockIdx.x * 8;
  const int f0 = blockIdx.y * 16;

  // float4 copies: 8*H = 2048 floats (2 iters), 16*H = 4096 floats (4 iters)
  for (int idx = tid * 4; idx < 8 * H; idx += 256 * 4) {
    int r = idx >> 8, c = idx & (H - 1);
    *(float4*)&piS[idx] = *(const float4*)&pi[(size_t)(i0 + r) * H + c];
  }
  for (int idx = tid * 4; idx < 16 * H; idx += 256 * 4) {
    int r = idx >> 8, c = idx & (H - 1);
    *(float4*)&pfS[idx] = *(const float4*)&pf[(size_t)(f0 + r) * H + c];
  }
  if (tid < H) b1S[tid] = b1[tid];
  if (tid < 128) { b2S[tid] = b2[tid]; w3S[tid] = w3[tid]; }
  __syncthreads();

  const float* piv = &piS[wave * H];   // this wave's intent row
  const float* pfv = &pfS[l16 * H];    // A-matrix row M = l16 = fix index
  const int kb = hi16 ? 8 : 0;

  v8f acc[8] = {};

  for (int k0 = 0; k0 < H; k0 += TK) {
    stage_w_tile(Wt2, w2, 128, k0, 0, tid);
    __syncthreads();

    // build A fragment = relu(pi + pf + b1) in bf16, per CDNA5 A lane layout
    v16bf afrag;
#pragma unroll
    for (int t = 0; t < 8; ++t) {
      int k = k0 + kb + t;
      float v  = piv[k]      + pfv[k]      + b1S[k];
      float v2 = piv[k + 16] + pfv[k + 16] + b1S[k + 16];
      afrag[t]     = (bf16)fmaxf(v, 0.0f);
      afrag[t + 8] = (bf16)fmaxf(v2, 0.0f);
    }

#pragma unroll
    for (int j = 0; j < 8; ++j) {
      const bf16* bcol = &Wt2[(j * 16 + l16) * TK + (hi16 ? 8 : 0)];
      v8bf blo = *(const v8bf*)(bcol);
      v8bf bhi = *(const v8bf*)(bcol + 16);
      v16bf bfrag = __builtin_shufflevector(blo, bhi,
          0,1,2,3,4,5,6,7,8,9,10,11,12,13,14,15);
      acc[j] = __builtin_amdgcn_wmma_f32_16x16x32_bf16(
          false, afrag, false, bfrag, (short)0, acc[j], false, false);
    }
    __syncthreads();
  }

  // epilogue: h2 = relu(acc + b2); rowsum = h2 . w3; butterfly-reduce 16 lanes
  float partial[8];
#pragma unroll
  for (int r = 0; r < 8; ++r) partial[r] = 0.0f;
#pragma unroll
  for (int j = 0; j < 8; ++j) {
    int n = j * 16 + l16;
    float bb = b2S[n], ww = w3S[n];
#pragma unroll
    for (int r = 0; r < 8; ++r)
      partial[r] += fmaxf(acc[j][r] + bb, 0.0f) * ww;
  }
#pragma unroll
  for (int r = 0; r < 8; ++r)
    for (int off = 8; off > 0; off >>= 1)
      partial[r] += __shfl_xor(partial[r], off);

  if (l16 == 0) {
    float bb3 = b3[0];
#pragma unroll
    for (int r = 0; r < 8; ++r) {
      float s = partial[r] + bb3;                // row M = r + hi16*8 = fix
      out[(size_t)(i0 + wave) * NF + f0 + hi16 * 8 + r] =
          1.0f / (1.0f + __expf(-s));
    }
  }
}

// ---------------------------------------------------------------------------
// Host orchestration. Input order: setup_inputs() insertion order with params
// expanded as (w, b) pairs in their insertion order.
// ---------------------------------------------------------------------------
extern "C" void kernel_launch(void* const* d_in, const int* in_sizes, int n_in,
                              void* d_out, int out_size, void* d_ws, size_t ws_size,
                              hipStream_t stream)
{
  (void)in_sizes; (void)n_in; (void)out_size; (void)ws_size;

  const float* x_dealer = (const float*)d_in[0];
  // d_in[1] x_intent, d_in[2] x_fix: dead (hi/hf never reach the output)
  const float* enc_d_w = (const float*)d_in[3];
  const float* enc_d_b = (const float*)d_in[4];
  // d_in[5..8]: enc_intent / enc_fix (dead)
  const float* c1hi_w = (const float*)d_in[9];
  const float* c1hi_b = (const float*)d_in[10];
  // d_in[11..14]: c1_rb / c1_af (dead; f1 is unused)
  const float* c2hi_w = (const float*)d_in[15];
  const float* c2hi_b = (const float*)d_in[16];
  const float* c2rb_w = (const float*)d_in[17];
  const float* c2rb_b = (const float*)d_in[18];
  const float* c2af_w = (const float*)d_in[19];
  const float* c2af_b = (const float*)d_in[20];
  const float* p1_w = (const float*)d_in[21];
  const float* p1_b = (const float*)d_in[22];
  const float* p2_w = (const float*)d_in[23];
  const float* p2_b = (const float*)d_in[24];
  const float* p3_w = (const float*)d_in[25];
  const float* p3_b = (const float*)d_in[26];
  const int* ei_hi = (const int*)d_in[27];
  const int* ei_rb = (const int*)d_in[28];
  const int* ei_af = (const int*)d_in[29];

  float* ws = (float*)d_ws;
  size_t off = 0;
  auto carve = [&](size_t n) { float* p = ws + off; off += n; return p; };
  float* hd   = carve((size_t)ND * H);
  float* tmp  = carve((size_t)ND * H);
  float* aggI = carve((size_t)NI * H);
  float* aggF = carve((size_t)NF * H);
  float* i1   = carve((size_t)NI * H);
  float* i2   = carve((size_t)NI * H);
  float* f2   = carve((size_t)NF * H);
  float* piB  = carve((size_t)NI * H);
  float* pfB  = carve((size_t)NF * H);
  float* degs = carve(ND + NI + NI + NF + ND + NF);   // contiguous
  float* dShi = degs;
  float* dDhi = dShi + ND;
  float* dSrb = dDhi + NI;
  float* dDrb = dSrb + NI;
  float* dSaf = dDrb + NF;
  float* dDaf = dSaf + ND;
  const int NDEG = ND + NI + NI + NF + ND + NF;

  auto gemm = [&](const float* A, const float* W, const float* bias, float* C,
                  int M, int N, int K, int act) {
    dim3 g(M / TM, N / TN);
    gemm_bias_act_kernel<<<g, dim3(256), 0, stream>>>(A, W, bias, C, M, N, K, act);
  };
  auto scatter = [&](const float* feat, const int* ei, const float* iS,
                     const float* iD, float* dst, int E) {
    int threads = E * (H / 4);
    scatter_kernel<<<(threads + 255) / 256, 256, 0, stream>>>(feat, ei, iS, iD, dst, E);
  };
  auto finalize = [&](float* dst, const float* acc, const float* ba,
                      const float* bb, int n, int relu_) {
    finalize_kernel<<<(n + 255) / 256, 256, 0, stream>>>(dst, acc, ba, bb, n, relu_);
  };

  // --- degrees & symmetric normalization ---
  hipMemsetAsync(degs, 0, (size_t)NDEG * sizeof(float), stream);
  deg_kernel<<<(E_HI + 255) / 256, 256, 0, stream>>>(ei_hi, E_HI, dShi, dDhi);
  deg_kernel<<<(E_RB + 255) / 256, 256, 0, stream>>>(ei_rb, E_RB, dSrb, dDrb);
  deg_kernel<<<(E_AF + 255) / 256, 256, 0, stream>>>(ei_af, E_AF, dSaf, dDaf);
  invsqrt_kernel<<<(NDEG + 255) / 256, 256, 0, stream>>>(degs, NDEG);

  // --- dealer encoder: hd = relu(x_dealer @ We + be) ---
  gemm(x_dealer, enc_d_w, enc_d_b, hd, ND, H, DD, 1);

  // --- i1 = relu(gcn(hd, ei_hi, c1_hi)) ---
  gemm(hd, c1hi_w, nullptr, tmp, ND, H, H, 0);
  hipMemsetAsync(aggI, 0, (size_t)NI * H * sizeof(float), stream);
  scatter(tmp, ei_hi, dShi, dDhi, aggI, E_HI);
  finalize(i1, aggI, c1hi_b, nullptr, NI * H, 1);

  // --- i2 = gcn(hd, ei_hi, c2_hi) + b ---
  gemm(hd, c2hi_w, nullptr, tmp, ND, H, H, 0);
  hipMemsetAsync(aggI, 0, (size_t)NI * H * sizeof(float), stream);
  scatter(tmp, ei_hi, dShi, dDhi, aggI, E_HI);
  finalize(i2, aggI, c2hi_b, nullptr, NI * H, 0);

  // --- f2 = gcn(i1, ei_rb, c2_rb) + gcn(hd, ei_af, c2_af) + b_rb + b_af ---
  hipMemsetAsync(aggF, 0, (size_t)NF * H * sizeof(float), stream);
  gemm(i1, c2rb_w, nullptr, tmp, NI, H, H, 0);
  scatter(tmp, ei_rb, dSrb, dDrb, aggF, E_RB);
  gemm(hd, c2af_w, nullptr, tmp, ND, H, H, 0);
  scatter(tmp, ei_af, dSaf, dDaf, aggF, E_AF);
  finalize(f2, aggF, c2rb_b, c2af_b, NF * H, 0);

  // --- pi = i2 @ w1[:H], pf = f2 @ w1[H:] ---
  gemm(i2, p1_w, nullptr, piB, NI, H, H, 0);
  gemm(f2, p1_w + (size_t)H * H, nullptr, pfB, NF, H, H, 0);

  // --- fused cross-product MLP + sigmoid ---
  pair_mlp_kernel<<<dim3(NI / 8, NF / 16), dim3(256), 0, stream>>>(
      piB, pfB, p1_b, p2_w, p2_b, p3_w, p3_b, (float*)d_out);
}
